// TemporalMamba_32530082300158
// MI455X (gfx1250) — matrix operand order repaired and checked by reading, compiled, not run on hardware
//
#include <hip/hip_runtime.h>
#include <math.h>

// ---------------------------------------------------------------------------
// TemporalMamba forward for gfx1250 (MI455X). GEMMs via v_wmma_f32_16x16x32_f16.
// ---------------------------------------------------------------------------

typedef _Float16 half16 __attribute__((ext_vector_type(16)));
typedef _Float16 half4  __attribute__((ext_vector_type(4)));
typedef float    float8 __attribute__((ext_vector_type(8)));

#define NB      16          // batch
#define SEQ     512         // sequence length
#define TOK     (NB*SEQ)    // 8192 tokens
#define DMODEL  512
#define DIN     1024        // D_INNER
#define DSTATE  16
#define DTRANK  32
#define LDSS    40          // padded LDS row stride (halves); 80B rows keep 16B align

__device__ __forceinline__ float dev_silu(float x) { return x * (1.0f / (1.0f + __expf(-x))); }
__device__ __forceinline__ float dev_softplus(float x) { return (x > 20.f) ? x : log1pf(__expf(x)); }

// ---------------------------------------------------------------------------
// WMMA GEMM:  C[M,N] = epi( A[M,K] * W[N,K]^T + bias )
//   A row-major (lda), W row-major (rows = K-vectors), C row-major (ldc).
//   M % 128 == 0, N % 64 == 0, K % 32 == 0 at all call sites.
//   EPI: 0 = none, 1 = +bias, 2 = +bias then softplus
// Workgroup = 256 thr = 8 waves; tile 128x64; wave tile 32x32 (4 accumulators,
// 4 v_wmma per K-step). float4 global staging + f32->f16 convert, ds_store_b64,
// software-pipelined so next tile's loads overlap the WMMAs.
// ---------------------------------------------------------------------------
template<int EPI>
__global__ __launch_bounds__(256) void wmma_gemm(
    const float* __restrict__ A, int lda,
    const float* __restrict__ W,
    const float* __restrict__ bias,
    float* __restrict__ C, int ldc,
    int M, int N, int K)
{
  __shared__ _Float16 As[128 * LDSS];
  __shared__ _Float16 Ws[64 * LDSS];
  const int tid  = threadIdx.x;
  const int lane = tid & 31;
  const int wave = tid >> 5;
  const int wm   = wave >> 1;       // 0..3  -> M offset wm*32
  const int wn   = wave & 1;        // 0..1  -> N offset wn*32
  const int m0   = blockIdx.x * 128;
  const int n0   = blockIdx.y * 64;
  const int l15  = lane & 15;
  const int hi   = lane >> 4;       // half-wave select

  float8 acc00 = {0.f,0.f,0.f,0.f,0.f,0.f,0.f,0.f};
  float8 acc01 = acc00, acc10 = acc00, acc11 = acc00;

  // staging registers: A 128x32 -> 4 float4/thread, W 64x32 -> 2 float4/thread
  float4 ra[4];
  float4 rw[2];

  auto load_regs = [&](int k0) {
#pragma unroll
    for (int i = 0; i < 4; ++i) {
      int q = tid + i * 256;            // 1024 quads
      int r = q >> 3, c4 = (q & 7) * 4;
      ra[i] = *reinterpret_cast<const float4*>(&A[(size_t)(m0 + r) * lda + k0 + c4]);
    }
#pragma unroll
    for (int i = 0; i < 2; ++i) {
      int q = tid + i * 256;            // 512 quads
      int r = q >> 3, c4 = (q & 7) * 4;
      rw[i] = *reinterpret_cast<const float4*>(&W[(size_t)(n0 + r) * K + k0 + c4]);
    }
  };

  load_regs(0);

  for (int k0 = 0; k0 < K; k0 += 32) {
    // convert + store the staged tile to LDS
#pragma unroll
    for (int i = 0; i < 4; ++i) {
      int q = tid + i * 256;
      int r = q >> 3, c4 = (q & 7) * 4;
      half4 h;
      h[0] = (_Float16)ra[i].x; h[1] = (_Float16)ra[i].y;
      h[2] = (_Float16)ra[i].z; h[3] = (_Float16)ra[i].w;
      *reinterpret_cast<half4*>(&As[r * LDSS + c4]) = h;
    }
#pragma unroll
    for (int i = 0; i < 2; ++i) {
      int q = tid + i * 256;
      int r = q >> 3, c4 = (q & 7) * 4;
      half4 h;
      h[0] = (_Float16)rw[i].x; h[1] = (_Float16)rw[i].y;
      h[2] = (_Float16)rw[i].z; h[3] = (_Float16)rw[i].w;
      *reinterpret_cast<half4*>(&Ws[r * LDSS + c4]) = h;
    }
    __syncthreads();

    // issue next tile's global loads now; they stay outstanding across the WMMAs
    if (k0 + 32 < K) load_regs(k0 + 32);

    // A fragments (ISA 16-bit A 16x32 layout): two 16-row sub-tiles
    half16 a0{}, a1{};
    {
      int row0 = (wm * 32 + l15) * LDSS;
      int row1 = row0 + 16 * LDSS;
      int base = hi * 8;
#pragma unroll
      for (int p = 0; p < 8; ++p) {
        int koff = base + ((p & 3) * 2) + ((p & 4) ? 16 : 0);
        a0[2*p]   = As[row0 + koff];
        a0[2*p+1] = As[row0 + koff + 1];
        a1[2*p]   = As[row1 + koff];
        a1[2*p+1] = As[row1 + koff + 1];
      }
    }
    // B fragments: lane -> column N, K in table order (8h+s, 16+8h+s-8) per half-wave
    half16 b0{}, b1{};
    {
      int kb = hi * 8;
      int r0 = (wn * 32 + l15) * LDSS + kb;
      int r1 = r0 + 16 * LDSS;
#pragma unroll
      for (int s = 0; s < 16; ++s) {
        b0[s] = Ws[r0 + s + ((s & 8) ? 8 : 0)];
        b1[s] = Ws[r1 + s + ((s & 8) ? 8 : 0)];
      }
    }
    acc00 = __builtin_amdgcn_wmma_f32_16x16x32_f16(false, a0, false, b0, (short)0, acc00, false, false);
    acc01 = __builtin_amdgcn_wmma_f32_16x16x32_f16(false, a0, false, b1, (short)0, acc01, false, false);
    acc10 = __builtin_amdgcn_wmma_f32_16x16x32_f16(false, a1, false, b0, (short)0, acc10, false, false);
    acc11 = __builtin_amdgcn_wmma_f32_16x16x32_f16(false, a1, false, b1, (short)0, acc11, false, false);
    __syncthreads();
  }

  // epilogue + store (C/D layout: VGPR v -> M = v + hi*8, lane -> N)
#pragma unroll
  for (int mi = 0; mi < 2; ++mi) {
#pragma unroll
    for (int nt = 0; nt < 2; ++nt) {
      float8 acc = mi ? (nt ? acc11 : acc10) : (nt ? acc01 : acc00);
      int n = n0 + wn * 32 + nt * 16 + l15;
      float bv = (EPI >= 1) ? bias[n] : 0.f;
#pragma unroll
      for (int v = 0; v < 8; ++v) {
        int m = m0 + wm * 32 + mi * 16 + v + hi * 8;
        float val = acc[v] + bv;
        if (EPI == 2) val = dev_softplus(val);
        C[(size_t)m * ldc + n] = val;
      }
    }
  }
}

// ---------------------------------------------------------------------------
// Naive GEMM for small/awkward shapes (K=37, N=37, M=16). ACT: 0 none, 1 silu
// ---------------------------------------------------------------------------
template<int ACT>
__global__ __launch_bounds__(256) void naive_gemm(
    const float* __restrict__ A, int lda,
    const float* __restrict__ W,
    const float* __restrict__ bias,
    float* __restrict__ C, int ldc,
    int M, int N, int K)
{
  int idx = blockIdx.x * 256 + threadIdx.x;
  if (idx >= M * N) return;
  int m = idx / N, n = idx - m * N;
  float acc = bias ? bias[n] : 0.f;
  for (int k = 0; k < K; ++k)
    acc += A[(size_t)m * lda + k] * W[(size_t)n * K + k];
  if (ACT == 1) acc = dev_silu(acc);
  C[(size_t)m * ldc + n] = acc;
}

// ---------------------------------------------------------------------------
// LayerNorm over last dim (512). One block (256 thr) per token row.
// ---------------------------------------------------------------------------
__global__ __launch_bounds__(256) void ln_kernel(
    const float* __restrict__ x, const float* __restrict__ w,
    const float* __restrict__ b, float* __restrict__ y)
{
  __shared__ float sd[256];
  const int row = blockIdx.x;
  const int tid = threadIdx.x;
  const float* xr = x + (size_t)row * DMODEL;
  float a0 = xr[tid], a1 = xr[tid + 256];
  sd[tid] = a0 + a1; __syncthreads();
  for (int s = 128; s > 0; s >>= 1) { if (tid < s) sd[tid] += sd[tid + s]; __syncthreads(); }
  float mean = sd[0] * (1.f / DMODEL); __syncthreads();
  float d0 = a0 - mean, d1 = a1 - mean;
  sd[tid] = d0 * d0 + d1 * d1; __syncthreads();
  for (int s = 128; s > 0; s >>= 1) { if (tid < s) sd[tid] += sd[tid + s]; __syncthreads(); }
  float rstd = rsqrtf(sd[0] * (1.f / DMODEL) + 1e-5f);
  float* yr = y + (size_t)row * DMODEL;
  yr[tid]       = d0 * rstd * w[tid]       + b[tid];
  yr[tid + 256] = d1 * rstd * w[tid + 256] + b[tid + 256];
}

// ---------------------------------------------------------------------------
// Causal depthwise conv(4) + bias + SiLU. u lives in xz[:, 0:1024] (stride 2048)
// ---------------------------------------------------------------------------
__global__ __launch_bounds__(256) void conv_silu_kernel(
    const float* __restrict__ xz, const float* __restrict__ cw,
    const float* __restrict__ cb, float* __restrict__ out)
{
  size_t i = (size_t)blockIdx.x * 256 + threadIdx.x;
  if (i >= (size_t)TOK * DIN) return;
  int d = (int)(i & (DIN - 1));
  size_t t = i >> 10;
  int l = (int)(t & (SEQ - 1));
  float acc = cb[d];
#pragma unroll
  for (int j = 0; j < 4; ++j) {
    int lj = l - 3 + j;
    if (lj >= 0) acc += cw[d * 4 + j] * xz[(t - 3 + j) * 2048 + d];
  }
  out[i] = dev_silu(acc);
}

// ---------------------------------------------------------------------------
// Selective scan. grid = NB*4 blocks, 256 threads; thread = one (b, d) channel.
// Fuses +u*D and *silu(z). B_t/C_t broadcast through LDS.
// ---------------------------------------------------------------------------
__global__ __launch_bounds__(256) void scan_kernel(
    const float* __restrict__ dt,   // TOK x 1024 (post softplus)
    const float* __restrict__ xdbl, // TOK x 64 : [dt_r 0:32 | B 32:48 | C 48:64]
    const float* __restrict__ u,    // TOK x 1024 (post conv+silu)
    const float* __restrict__ xz,   // TOK x 2048 (z at +1024)
    const float* __restrict__ A_log,// 1024 x 16
    const float* __restrict__ Dv,   // 1024
    float* __restrict__ y)          // TOK x 1024
{
  const int b = blockIdx.x >> 2;
  const int d = ((blockIdx.x & 3) << 8) + threadIdx.x;
  float A[DSTATE], h[DSTATE];
#pragma unroll
  for (int j = 0; j < DSTATE; ++j) { A[j] = -__expf(A_log[d * DSTATE + j]); h[j] = 0.f; }
  const float Dd = Dv[d];
  __shared__ float Bs[DSTATE], Cs[DSTATE];
  for (int l = 0; l < SEQ; ++l) {
    size_t t = (size_t)b * SEQ + l;
    __syncthreads();
    if (threadIdx.x < 32) {
      int j = threadIdx.x & 15;
      if (threadIdx.x < 16) Bs[j] = xdbl[t * 64 + 32 + j];
      else                  Cs[j] = xdbl[t * 64 + 48 + j];
    }
    __syncthreads();
    float dtv = dt[t * DIN + d];
    float uv  = u[t * DIN + d];
    float du  = dtv * uv;
    float yv  = 0.f;
#pragma unroll
    for (int j = 0; j < DSTATE; ++j) {
      h[j] = h[j] * __expf(dtv * A[j]) + du * Bs[j];
      yv  += h[j] * Cs[j];
    }
    yv += uv * Dd;
    float z = xz[t * 2048 + DIN + d];
    y[t * DIN + d] = yv * dev_silu(z);
  }
}

// ---------------------------------------------------------------------------
// Elementwise helpers
// ---------------------------------------------------------------------------
__global__ __launch_bounds__(256) void add_kernel(float* __restrict__ res,
                                                  const float* __restrict__ h,
                                                  size_t n, int copy)
{
  size_t i = (size_t)blockIdx.x * 256 + threadIdx.x;
  if (i < n) res[i] = copy ? h[i] : res[i] + h[i];
}

__global__ __launch_bounds__(256) void gate_kernel(const float* __restrict__ in,
                                                   float* __restrict__ out)
{
  size_t i = (size_t)blockIdx.x * 256 + threadIdx.x;
  if (i >= (size_t)TOK * DIN) return;
  size_t t = i >> 10; int c = (int)(i & (DIN - 1));
  float a = in[t * 2048 + c];
  float g = in[t * 2048 + DIN + c];
  out[i] = a * dev_silu(g);
}

__global__ __launch_bounds__(256) void concat_kernel(const float* __restrict__ a,
                                                     const float* __restrict__ b,
                                                     float* __restrict__ out)
{
  size_t i = (size_t)blockIdx.x * 256 + threadIdx.x;
  if (i >= (size_t)TOK * DIN) return;
  size_t t = i >> 10; int c = (int)(i & (DIN - 1));
  out[i] = (c < DMODEL) ? a[t * DMODEL + c] : b[t * DMODEL + (c - DMODEL)];
}

__global__ __launch_bounds__(256) void add_temb_kernel(float* __restrict__ h,
                                                       const float* __restrict__ temb)
{
  size_t i = (size_t)blockIdx.x * 256 + threadIdx.x;
  if (i >= (size_t)TOK * DMODEL) return;
  size_t t = i >> 9;         // token index (DMODEL = 512)
  int    c = (int)(i & (DMODEL - 1));
  int    b = (int)(t >> 9);  // batch index (SEQ = 512)
  h[i] += temb[(size_t)b * DMODEL + c];
}

__global__ __launch_bounds__(256) void sin_emb_kernel(const float* __restrict__ time,
                                                      float* __restrict__ emb)
{
  int i = blockIdx.x * 256 + threadIdx.x;
  if (i >= NB * 256) return;
  int b = i >> 8, j = i & 255;
  float freq = __expf((float)j * (-logf(10000.f) / 255.f));
  float ang  = time[b] * freq;
  emb[(size_t)b * DMODEL + j]       = __sinf(ang);
  emb[(size_t)b * DMODEL + 256 + j] = __cosf(ang);
}

// ---------------------------------------------------------------------------
// Host orchestration
// ---------------------------------------------------------------------------
struct Blk {
  const float *norm_w, *norm_b;
  const float *in_proj, *conv_w, *conv_b, *x_proj, *dt_w, *dt_b, *A_log, *D, *out_proj;
  const float *norm2_w, *norm2_b, *fc1_w, *fc1_b, *fc2_w, *fc2_b;
  const float *skip_w, *skip_b;
};

static inline dim3 ewg(size_t n) { return dim3((unsigned)((n + 255) / 256)); }

static void run_block(hipStream_t s, const Blk& bp, const float* skip, bool first,
                      float* hbuf, float* resbuf, float* xz, float* ucv,
                      float* dtb, float* ysc, float* xdbl, float* catb)
{
  const size_t nTD = (size_t)TOK * DMODEL;
  const size_t nTI = (size_t)TOK * DIN;
  const dim3 gM(TOK / 128, 1);
  if (skip) {
    concat_kernel<<<ewg(nTI), 256, 0, s>>>(hbuf, skip, catb);
    wmma_gemm<1><<<dim3(TOK/128, DMODEL/64), 256, 0, s>>>(catb, DIN, bp.skip_w, bp.skip_b,
                                                          hbuf, DMODEL, TOK, DMODEL, DIN);
  }
  add_kernel<<<ewg(nTD), 256, 0, s>>>(resbuf, hbuf, nTD, first ? 1 : 0);
  ln_kernel<<<TOK, 256, 0, s>>>(resbuf, bp.norm_w, bp.norm_b, hbuf);
  // ---- Mamba mixer ----
  wmma_gemm<0><<<dim3(TOK/128, 2048/64), 256, 0, s>>>(hbuf, DMODEL, bp.in_proj, nullptr,
                                                      xz, 2048, TOK, 2048, DMODEL);
  conv_silu_kernel<<<ewg(nTI), 256, 0, s>>>(xz, bp.conv_w, bp.conv_b, ucv);
  wmma_gemm<0><<<dim3(TOK/128, 64/64), 256, 0, s>>>(ucv, DIN, bp.x_proj, nullptr,
                                                    xdbl, 64, TOK, 64, DIN);
  wmma_gemm<2><<<dim3(TOK/128, DIN/64), 256, 0, s>>>(xdbl, 64, bp.dt_w, bp.dt_b,
                                                     dtb, DIN, TOK, DIN, DTRANK);
  scan_kernel<<<NB * 4, 256, 0, s>>>(dtb, xdbl, ucv, xz, bp.A_log, bp.D, ysc);
  wmma_gemm<0><<<dim3(TOK/128, DMODEL/64), 256, 0, s>>>(ysc, DIN, bp.out_proj, nullptr,
                                                        hbuf, DMODEL, TOK, DMODEL, DIN);
  add_kernel<<<ewg(nTD), 256, 0, s>>>(resbuf, hbuf, nTD, 0);
  ln_kernel<<<TOK, 256, 0, s>>>(resbuf, bp.norm2_w, bp.norm2_b, hbuf);
  // ---- Gated MLP ----
  wmma_gemm<1><<<dim3(TOK/128, 2048/64), 256, 0, s>>>(hbuf, DMODEL, bp.fc1_w, bp.fc1_b,
                                                      xz, 2048, TOK, 2048, DMODEL);
  gate_kernel<<<ewg(nTI), 256, 0, s>>>(xz, ucv);
  wmma_gemm<1><<<dim3(TOK/128, DMODEL/64), 256, 0, s>>>(ucv, DIN, bp.fc2_w, bp.fc2_b,
                                                        hbuf, DMODEL, TOK, DMODEL, DIN);
  (void)gM;
}

extern "C" void kernel_launch(void* const* d_in, const int* in_sizes, int n_in,
                              void* d_out, int out_size, void* d_ws, size_t ws_size,
                              hipStream_t stream)
{
  (void)in_sizes; (void)n_in; (void)out_size; (void)ws_size;
  const float* x    = (const float*)d_in[0];
  /* cond (d_in[1]) unused by the reference forward */
  const float* time = (const float*)d_in[2];
  int idx = 3;
  auto F = [&](void) { return (const float*)d_in[idx++]; };
  const float* proj_up_w = F(); const float* proj_up_b = F();
  const float* proj_dn_w = F(); const float* proj_dn_b = F();
  const float* tm1_w = F();     const float* tm1_b = F();
  const float* tm2_w = F();     const float* tm2_b = F();

  Blk blocks[7];
  for (int i = 0; i < 7; ++i) {
    bool has_skip = (i >= 4);
    Blk& b = blocks[i];
    b.norm_w = F(); b.norm_b = F();
    b.in_proj = F(); b.conv_w = F(); b.conv_b = F(); b.x_proj = F();
    b.dt_w = F(); b.dt_b = F(); b.A_log = F(); b.D = F(); b.out_proj = F();
    b.norm2_w = F(); b.norm2_b = F();
    b.fc1_w = F(); b.fc1_b = F(); b.fc2_w = F(); b.fc2_b = F();
    if (has_skip) { b.skip_w = F(); b.skip_b = F(); }
    else          { b.skip_w = nullptr; b.skip_b = nullptr; }
  }

  // workspace layout (floats)
  float* ws = (float*)d_ws;
  size_t o = 0;
  float* hbuf   = ws + o; o += (size_t)TOK * DMODEL;
  float* resbuf = ws + o; o += (size_t)TOK * DMODEL;
  float* skipb0 = ws + o; o += (size_t)TOK * DMODEL;
  float* skipb1 = ws + o; o += (size_t)TOK * DMODEL;
  float* skipb2 = ws + o; o += (size_t)TOK * DMODEL;
  float* xz     = ws + o; o += (size_t)TOK * 2048;
  float* ucv    = ws + o; o += (size_t)TOK * DIN;
  float* dtb    = ws + o; o += (size_t)TOK * DIN;
  float* ysc    = ws + o; o += (size_t)TOK * DIN;
  float* xdbl   = ws + o; o += (size_t)TOK * 64;
  float* catb   = ws + o; o += (size_t)TOK * DIN;
  float* tembA  = ws + o; o += (size_t)NB * DMODEL;
  float* tembB  = ws + o; o += (size_t)NB * 2048;
  float* tembC  = ws + o; o += (size_t)NB * DMODEL;

  const size_t nTD = (size_t)TOK * DMODEL;

  // h = proj_up(x)
  naive_gemm<0><<<ewg((size_t)TOK * DMODEL), 256, 0, stream>>>(
      x, 37, proj_up_w, proj_up_b, hbuf, DMODEL, TOK, DMODEL, 37);
  // time embedding MLP
  sin_emb_kernel<<<ewg((size_t)NB * 256), 256, 0, stream>>>(time, tembA);
  naive_gemm<1><<<ewg((size_t)NB * 2048), 256, 0, stream>>>(
      tembA, DMODEL, tm1_w, tm1_b, tembB, 2048, NB, 2048, DMODEL);
  naive_gemm<0><<<ewg((size_t)NB * DMODEL), 256, 0, stream>>>(
      tembB, 2048, tm2_w, tm2_b, tembC, DMODEL, NB, DMODEL, 2048);
  add_temb_kernel<<<ewg(nTD), 256, 0, stream>>>(hbuf, tembC);

  // down blocks (push skips)
  float* skips[3] = { skipb0, skipb1, skipb2 };
  for (int i = 0; i < 3; ++i) {
    run_block(stream, blocks[i], nullptr, i == 0, hbuf, resbuf, xz, ucv, dtb, ysc, xdbl, catb);
    hipMemcpyAsync(skips[i], hbuf, nTD * sizeof(float), hipMemcpyDeviceToDevice, stream);
  }
  // mid
  run_block(stream, blocks[3], nullptr, false, hbuf, resbuf, xz, ucv, dtb, ysc, xdbl, catb);
  // up blocks (pop skips in LIFO order)
  for (int i = 0; i < 3; ++i) {
    run_block(stream, blocks[4 + i], skips[2 - i], false, hbuf, resbuf, xz, ucv, dtb, ysc, xdbl, catb);
  }
  // out = proj_down(h)
  naive_gemm<0><<<ewg((size_t)TOK * 37), 256, 0, stream>>>(
      hbuf, DMODEL, proj_dn_w, proj_dn_b, (float*)d_out, 37, TOK, 37, DMODEL);
}
